// MultiHeadAttention_74354473828481
// MI455X (gfx1250) — matrix operand order repaired and checked by reading, not compiled, not run on hardware
//
#include <hip/hip_runtime.h>
#include <hip/hip_bf16.h>
#include <stdint.h>

typedef __bf16 v16bf __attribute__((ext_vector_type(16)));
typedef float  v8f   __attribute__((ext_vector_type(8)));
typedef int    v4i_  __attribute__((ext_vector_type(4)));

#define B_  2
#define S_  2048
#define D_  1024
#define H_  16
#define HD_ 64
#define MTOT_ (B_*S_)   // 4096

#define AS1_ __attribute__((address_space(1)))
#define AS3_ __attribute__((address_space(3)))

#if defined(__has_builtin)
#  if __has_builtin(__builtin_amdgcn_global_load_async_to_lds_b128)
#    define HAVE_ASYNC_LDS 1
#  endif
#endif
#ifndef HAVE_ASYNC_LDS
#  define HAVE_ASYNC_LDS 0
#endif

union FragBF { v16bf v; unsigned int u[8]; };

__device__ __forceinline__ unsigned short f2bf(float f) {
  union { float f; unsigned u; } x; x.f = f;
  unsigned r = x.u + 0x7fffu + ((x.u >> 16) & 1u);
  return (unsigned short)(r >> 16);
}

__device__ __forceinline__ void wait_async0() {
#if defined(__has_builtin) && __has_builtin(__builtin_amdgcn_s_wait_asynccnt)
  __builtin_amdgcn_s_wait_asynccnt(0);
#else
  asm volatile("s_wait_asynccnt 0" ::: "memory");
#endif
}

#if HAVE_ASYNC_LDS
__device__ __forceinline__ void async_b128(const void* g, void* l) {
  __builtin_amdgcn_global_load_async_to_lds_b128(
      (AS1_ v4i_*)g, (AS3_ v4i_*)l, 0, 0);
}
#endif

// ---------------------------------------------------------------- convert ---
__global__ void convert_f32_to_bf16(const float* __restrict__ src,
                                    unsigned short* __restrict__ dst, int n) {
  int i = blockIdx.x * blockDim.x + threadIdx.x;
  int stride = gridDim.x * blockDim.x;
  for (; i < n; i += stride) dst[i] = f2bf(src[i]);
}

// ------------------------------------------------------------------- GEMM ---
// out[m,n] = sum_k A[m,k] * W[n,k] + bias[n]
// MODE 0: write bf16 head-split [B,H,S,HD]; MODE 1: write fp32 [M,N]
#define TILEHALVES (128 * 40)   // one LDS tile (padded) in bf16 halves

__device__ __forceinline__ void issue_tile_async(
    const unsigned short* __restrict__ A, const unsigned short* __restrict__ W,
    unsigned short* sA, unsigned short* sB, int m0, int n0, int tid, int kt,
    int buf) {
#if HAVE_ASYNC_LDS
  #pragma unroll
  for (int i = 0; i < 2; i++) {
    int idx = tid + i * 256;                 // 0..511
    int row = idx >> 2, c4 = idx & 3;        // 128 rows x 4 chunks of 8 bf16
    async_b128(A + (size_t)(m0 + row) * D_ + kt * 32 + c4 * 8,
               sA + buf * TILEHALVES + row * 40 + c4 * 8);
    async_b128(W + (size_t)(n0 + row) * D_ + kt * 32 + c4 * 8,
               sB + buf * TILEHALVES + row * 40 + c4 * 8);
  }
#endif
}

template <int MODE>
__global__ __launch_bounds__(256) void gemm_bf16_wmma(
    const unsigned short* __restrict__ A,   // [M,K] bf16 row-major
    const unsigned short* __restrict__ W,   // [N,K] bf16 row-major
    const float* __restrict__ bias,         // [N] fp32
    unsigned short* __restrict__ out_bf16,
    float* __restrict__ out_f32) {
  const int K = D_;
  __shared__ __align__(16) unsigned short sA[2 * TILEHALVES];
  __shared__ __align__(16) unsigned short sB[2 * TILEHALVES];

  const int tid = threadIdx.x, lane = tid & 31, wave = tid >> 5;
  const int wm = wave >> 1, wn = wave & 1;             // 4 x 2 wave grid
  const int m0 = blockIdx.x * 128, n0 = blockIdx.y * 128;
  const int lh = lane >> 4;                            // lane half (0/1)
  const int ll = lane & 15;

  v8f acc[2][4];
  #pragma unroll
  for (int i = 0; i < 2; i++)
    #pragma unroll
    for (int j = 0; j < 4; j++)
      #pragma unroll
      for (int e = 0; e < 8; e++) acc[i][j][e] = 0.0f;

#if HAVE_ASYNC_LDS
  // ---- double-buffered: async LDS fill of tile kt+1 overlaps WMMA of kt ----
  issue_tile_async(A, W, sA, sB, m0, n0, tid, 0, 0);
  #pragma unroll 1
  for (int kt = 0; kt < K / 32; ++kt) {
    const int cur = kt & 1;
    wait_async0();
    __syncthreads();
    if (kt + 1 < K / 32)
      issue_tile_async(A, W, sA, sB, m0, n0, tid, kt + 1, 1 - cur);
    const unsigned short* tA = sA + cur * TILEHALVES;
    const unsigned short* tB = sB + cur * TILEHALVES;
#else
  #pragma unroll 1
  for (int kt = 0; kt < K / 32; ++kt) {
    __syncthreads();
    #pragma unroll
    for (int i = 0; i < 2; i++) {
      int idx = tid + i * 256;
      int row = idx >> 2, c4 = idx & 3;
      *(uint4*)(sA + row * 40 + c4 * 8) =
          *(const uint4*)(A + (size_t)(m0 + row) * K + kt * 32 + c4 * 8);
      *(uint4*)(sB + row * 40 + c4 * 8) =
          *(const uint4*)(W + (size_t)(n0 + row) * K + kt * 32 + c4 * 8);
    }
    __syncthreads();
    const unsigned short* tA = sA;
    const unsigned short* tB = sB;
#endif
    FragBF aF[2], bF[4];
    #pragma unroll
    for (int mt = 0; mt < 2; ++mt) {
      int row = wm * 32 + mt * 16 + ll;
      #pragma unroll
      for (int p = 0; p < 8; p++) {
        int kk = (lh << 4) + p * 2;
        aF[mt].u[p] = *(const unsigned int*)(tA + row * 40 + kk);
      }
    }
    #pragma unroll
    for (int nt = 0; nt < 4; ++nt) {
      int col = wn * 64 + nt * 16 + ll;
      #pragma unroll
      for (int p = 0; p < 8; p++) {
        int kk = (lh << 4) + p * 2;
        bF[nt].u[p] = *(const unsigned int*)(tB + col * 40 + kk);
      }
    }
    #pragma unroll
    for (int mt = 0; mt < 2; ++mt)
      #pragma unroll
      for (int nt = 0; nt < 4; ++nt)
        acc[mt][nt] = __builtin_amdgcn_wmma_f32_16x16x32_bf16(
            false, aF[mt].v, false, bF[nt].v, (short)0, acc[mt][nt],
            false, false);
  }

  // epilogue (compile-time MODE: no per-element branching)
  #pragma unroll
  for (int mt = 0; mt < 2; ++mt)
    #pragma unroll
    for (int nt = 0; nt < 4; ++nt)
      #pragma unroll
      for (int e = 0; e < 8; e++) {
        int m = m0 + wm * 32 + mt * 16 + e + (lh << 3);
        int n = n0 + wn * 64 + nt * 16 + ll;
        float val = acc[mt][nt][e] + bias[n];
        if (MODE == 0) {
          int bb = m >> 11, ss = m & (S_ - 1);
          int hh = n >> 6,  hd = n & (HD_ - 1);
          out_bf16[(((size_t)bb * H_ + hh) * S_ + ss) * HD_ + hd] = f2bf(val);
        } else {
          out_f32[(size_t)m * D_ + n] = val;
        }
      }
}

// -------------------------------------------------------- flash attention ---
// Q/K/V in [B,H,S,HD] bf16; mask [B,1,S,S] int; ctx out [B,S,D] bf16.
// WG: 32 q-rows shared by 8 waves; wave w handles head blockIdx.z*8+w.
__global__ __launch_bounds__(256) void attn_fwd_wmma(
    const unsigned short* __restrict__ Qh,
    const unsigned short* __restrict__ Kh,
    const unsigned short* __restrict__ Vh,
    const int* __restrict__ mask,
    unsigned short* __restrict__ ctx) {
  __shared__ float sBias[32 * 32];
  __shared__ __align__(16) unsigned short sV[8 * 32 * 72];
  __shared__ __align__(16) unsigned short sP[8 * 16 * 36];

  const int tid = threadIdx.x, lane = tid & 31, wave = tid >> 5;
  const int lh = lane >> 4, ll = lane & 15;
  const int q0 = blockIdx.x * 32;
  const int b  = blockIdx.y;
  const int h  = blockIdx.z * 8 + wave;
  const int bh = b * H_ + h;
  unsigned short* sVw = sV + wave * (32 * 72);
  unsigned short* sPw = sP + wave * (16 * 36);

  // Q fragments: 2 m-tiles x 2 k-frags (HD=64 -> 2 x K32)
  FragBF qa[2][2];
  #pragma unroll
  for (int mt = 0; mt < 2; mt++)
    #pragma unroll
    for (int kf = 0; kf < 2; kf++)
      #pragma unroll
      for (int p = 0; p < 8; p++) {
        int row = q0 + mt * 16 + ll;
        int kk  = kf * 32 + (lh << 4) + p * 2;
        qa[mt][kf].u[p] =
            *(const unsigned int*)(Qh + (size_t)(bh * S_ + row) * HD_ + kk);
      }

  v8f o[2][4];
  float mrow[2][8], lrow[2][8];
  #pragma unroll
  for (int mt = 0; mt < 2; mt++) {
    #pragma unroll
    for (int e = 0; e < 8; e++) { mrow[mt][e] = -3.0e38f; lrow[mt][e] = 0.0f; }
    #pragma unroll
    for (int nt = 0; nt < 4; nt++)
      #pragma unroll
      for (int e = 0; e < 8; e++) o[mt][nt][e] = 0.0f;
  }

  const float scale = 0.125f;  // HD^-0.5

  #pragma unroll 1
  for (int kk0 = 0; kk0 < S_; kk0 += 32) {
    __syncthreads();
    // shared mask tile (reused by all 8 heads in the WG)
    #pragma unroll
    for (int i = 0; i < 4; i++) {
      int idx = tid + i * 256;
      int ql = idx >> 5, kl = idx & 31;
      int mv = mask[(size_t)b * S_ * S_ + (size_t)(q0 + ql) * S_ + kk0 + kl];
      sBias[idx] = (mv == 0) ? -__builtin_inff() : 0.0f;
    }
    // per-wave V tile (32 keys x 64 hd): async direct-to-LDS when available
#if HAVE_ASYNC_LDS
    #pragma unroll
    for (int i = 0; i < 8; i++) {
      int idx = lane + i * 32;
      int row = idx >> 3, c4 = idx & 7;
      async_b128(Vh + (size_t)(bh * S_ + kk0 + row) * HD_ + c4 * 8,
                 sVw + row * 72 + c4 * 8);
    }
    wait_async0();
#else
    #pragma unroll
    for (int i = 0; i < 8; i++) {
      int idx = lane + i * 32;
      int row = idx >> 3, c4 = idx & 7;
      *(uint4*)(sVw + row * 72 + c4 * 8) =
          *(const uint4*)(Vh + (size_t)(bh * S_ + kk0 + row) * HD_ + c4 * 8);
    }
#endif
    if (kk0 + 32 < S_)
      __builtin_prefetch(Kh + (size_t)(bh * S_ + kk0 + 32) * HD_, 0, 1);
    __syncthreads();

    // K^T fragments: B-matrix, N=key, contraction=hd
    FragBF kb[2][2];
    #pragma unroll
    for (int kt2 = 0; kt2 < 2; kt2++)
      #pragma unroll
      for (int kf = 0; kf < 2; kf++)
        #pragma unroll
        for (int p = 0; p < 8; p++) {
          int key = kk0 + kt2 * 16 + ll;
          int kd  = kf * 32 + (lh << 4) + p * 2;
          kb[kt2][kf].u[p] =
              *(const unsigned int*)(Kh + (size_t)(bh * S_ + key) * HD_ + kd);
        }
    // V fragments from LDS: B-matrix, N=hd, contraction=key
    FragBF vb[4];
    #pragma unroll
    for (int nt = 0; nt < 4; nt++)
      #pragma unroll
      for (int p = 0; p < 8; p++) {
        int col = nt * 16 + ll;
        int kr  = (lh << 4) + p * 2;
        unsigned lo = sVw[kr * 72 + col];
        unsigned hi = sVw[(kr + 1) * 72 + col];
        vb[nt].u[p] = lo | (hi << 16);
      }

    #pragma unroll
    for (int mt = 0; mt < 2; mt++) {
      v8f c0, c1;
      #pragma unroll
      for (int e = 0; e < 8; e++) { c0[e] = 0.0f; c1[e] = 0.0f; }
      c0 = __builtin_amdgcn_wmma_f32_16x16x32_bf16(false, qa[mt][0].v, false,
                kb[0][0].v, (short)0, c0, false, false);
      c0 = __builtin_amdgcn_wmma_f32_16x16x32_bf16(false, qa[mt][1].v, false,
                kb[0][1].v, (short)0, c0, false, false);
      c1 = __builtin_amdgcn_wmma_f32_16x16x32_bf16(false, qa[mt][0].v, false,
                kb[1][0].v, (short)0, c1, false, false);
      c1 = __builtin_amdgcn_wmma_f32_16x16x32_bf16(false, qa[mt][1].v, false,
                kb[1][1].v, (short)0, c1, false, false);

      float s0[8], s1[8], rmax[8], rsum[8], corr[8], mn[8], p0[8], p1[8];
      #pragma unroll
      for (int e = 0; e < 8; e++) {
        int M = e + (lh << 3);
        s0[e] = c0[e] * scale + sBias[(mt * 16 + M) * 32 + ll];
        s1[e] = c1[e] * scale + sBias[(mt * 16 + M) * 32 + 16 + ll];
        rmax[e] = fmaxf(s0[e], s1[e]);
      }
      #pragma unroll
      for (int e = 0; e < 8; e++)
        #pragma unroll
        for (int xm = 1; xm < 16; xm <<= 1)
          rmax[e] = fmaxf(rmax[e], __shfl_xor(rmax[e], xm, 32));
      #pragma unroll
      for (int e = 0; e < 8; e++) {
        mn[e]   = fmaxf(mrow[mt][e], rmax[e]);
        corr[e] = __expf(mrow[mt][e] - mn[e]);
        mrow[mt][e] = mn[e];
        p0[e] = __expf(s0[e] - mn[e]);
        p1[e] = __expf(s1[e] - mn[e]);
        rsum[e] = p0[e] + p1[e];
      }
      #pragma unroll
      for (int e = 0; e < 8; e++)
        #pragma unroll
        for (int xm = 1; xm < 16; xm <<= 1)
          rsum[e] += __shfl_xor(rsum[e], xm, 32);
      #pragma unroll
      for (int e = 0; e < 8; e++)
        lrow[mt][e] = lrow[mt][e] * corr[e] + rsum[e];
      #pragma unroll
      for (int nt = 0; nt < 4; nt++)
        #pragma unroll
        for (int e = 0; e < 8; e++) o[mt][nt][e] *= corr[e];

      // stage P (16x32) bf16 through per-wave LDS to build an A fragment
      #pragma unroll
      for (int e = 0; e < 8; e++) {
        int M = e + (lh << 3);
        sPw[M * 36 + ll]      = f2bf(p0[e]);
        sPw[M * 36 + 16 + ll] = f2bf(p1[e]);
      }
      asm volatile("s_wait_dscnt 0" ::: "memory");  // wave-local LDS fence
      FragBF pa;
      #pragma unroll
      for (int p = 0; p < 8; p++) {
        int kk = (lh << 4) + p * 2;
        pa.u[p] = *(const unsigned int*)(sPw + (ll) * 36 + kk);
      }
      #pragma unroll
      for (int nt = 0; nt < 4; nt++)
        o[mt][nt] = __builtin_amdgcn_wmma_f32_16x16x32_bf16(
            false, pa.v, false, vb[nt].v, (short)0, o[mt][nt], false, false);
    }
  }

  // epilogue: normalize and scatter context to [B,S,D] bf16
  #pragma unroll
  for (int mt = 0; mt < 2; mt++) {
    float inv[8];
    #pragma unroll
    for (int e = 0; e < 8; e++) inv[e] = 1.0f / lrow[mt][e];
    #pragma unroll
    for (int nt = 0; nt < 4; nt++)
      #pragma unroll
      for (int e = 0; e < 8; e++) {
        int ss  = q0 + mt * 16 + e + (lh << 3);
        int col = h * HD_ + nt * 16 + ll;
        ctx[((size_t)b * S_ + ss) * D_ + col] = f2bf(o[mt][nt][e] * inv[e]);
      }
  }
}

// ----------------------------------------------------------------- launch ---
extern "C" void kernel_launch(void* const* d_in, const int* in_sizes, int n_in,
                              void* d_out, int out_size, void* d_ws,
                              size_t ws_size, hipStream_t stream) {
  const float* q_in  = (const float*)d_in[0];
  const float* k_in  = (const float*)d_in[1];
  const float* v_in  = (const float*)d_in[2];
  const int*   amask = (const int*)  d_in[3];
  const float* Wq = (const float*)d_in[4];  const float* bq = (const float*)d_in[5];
  const float* Wk = (const float*)d_in[6];  const float* bk = (const float*)d_in[7];
  const float* Wv = (const float*)d_in[8];  const float* bv = (const float*)d_in[9];
  const float* Wo = (const float*)d_in[10]; const float* bo = (const float*)d_in[11];

  char* ws = (char*)d_ws;
  const size_t XB = (size_t)MTOT_ * D_ * 2;  // 8 MB (bf16 activation)
  const size_t WB = (size_t)D_ * D_ * 2;     // 2 MB (bf16 weight)
  unsigned short* Xq  = (unsigned short*)(ws + 0 * XB);
  unsigned short* Xk  = (unsigned short*)(ws + 1 * XB);
  unsigned short* Xv  = (unsigned short*)(ws + 2 * XB);
  unsigned short* Wqb = (unsigned short*)(ws + 3 * XB + 0 * WB);
  unsigned short* Wkb = (unsigned short*)(ws + 3 * XB + 1 * WB);
  unsigned short* Wvb = (unsigned short*)(ws + 3 * XB + 2 * WB);
  unsigned short* Wob = (unsigned short*)(ws + 3 * XB + 3 * WB);
  unsigned short* Qh  = (unsigned short*)(ws + 3 * XB + 4 * WB + 0 * XB);
  unsigned short* Kh  = (unsigned short*)(ws + 3 * XB + 4 * WB + 1 * XB);
  unsigned short* Vh  = (unsigned short*)(ws + 3 * XB + 4 * WB + 2 * XB);
  unsigned short* Ctx = (unsigned short*)(ws + 3 * XB + 4 * WB + 3 * XB);

  const int nAct = MTOT_ * D_;   // 4,194,304
  const int nW   = D_ * D_;      // 1,048,576
  convert_f32_to_bf16<<<2048, 256, 0, stream>>>(q_in, Xq, nAct);
  convert_f32_to_bf16<<<2048, 256, 0, stream>>>(k_in, Xk, nAct);
  convert_f32_to_bf16<<<2048, 256, 0, stream>>>(v_in, Xv, nAct);
  convert_f32_to_bf16<<<1024, 256, 0, stream>>>(Wq, Wqb, nW);
  convert_f32_to_bf16<<<1024, 256, 0, stream>>>(Wk, Wkb, nW);
  convert_f32_to_bf16<<<1024, 256, 0, stream>>>(Wv, Wvb, nW);
  convert_f32_to_bf16<<<1024, 256, 0, stream>>>(Wo, Wob, nW);

  dim3 ggrid(MTOT_ / 128, D_ / 128);
  gemm_bf16_wmma<0><<<ggrid, 256, 0, stream>>>(Xq, Wqb, bq, Qh, nullptr);
  gemm_bf16_wmma<0><<<ggrid, 256, 0, stream>>>(Xk, Wkb, bk, Kh, nullptr);
  gemm_bf16_wmma<0><<<ggrid, 256, 0, stream>>>(Xv, Wvb, bv, Vh, nullptr);

  attn_fwd_wmma<<<dim3(S_ / 32, B_, 2), 256, 0, stream>>>(Qh, Kh, Vh, amask, Ctx);

  gemm_bf16_wmma<1><<<ggrid, 256, 0, stream>>>(Ctx, Wob, bo, nullptr,
                                               (float*)d_out);
}
